// agcn_61366492725531
// MI455X (gfx1250) — matrix operand order repaired and checked by reading, not compiled, not run on hardware
//
#include <hip/hip_runtime.h>
#include <hip/hip_bf16.h>
#include <math.h>

// Problem constants (from reference)
#define N_  64
#define T_  300
#define V_  25
#define C_  64
#define K_  3
#define F_  64
#define I_  16
// D = T*I = 4800, scale = 1/sqrt(4800)
#define SCALE_ 0.014433756729740643f

// t-loop split for the attention kernel
#define CH_  10          // chunks
#define TCH_ 30          // t per chunk

// ---- workspace layout ----
// floats: Sws[K_][N_][32*32] (logits) then Cws[K_][N_][32] (column bias term)
#define SWS_FLOATS  (K_ * N_ * 32 * 32)          // 196608
#define CWS_FLOATS  (K_ * N_ * 32)               // 6144
#define ZERO_FLOATS (SWS_FLOATS + CWS_FLOATS)    // 202752 = 198*256*4
// bf16 pack region starts after the float region (offsets in bf16 elements):
#define WPACK_OFF   0          // 24 B-frags of W (nt 0..11 x kc 0..1)
#define WQPACK_OFF  12288      // 6 A-frags of Wq^T (k x kc)
#define WKPACK_OFF  15360      // 6 A-frags of Wk^T
#define APACK_OFF   18432      // 192 (k,n) x 2 mtiles A-frags of A_adapt^T

typedef __attribute__((ext_vector_type(16))) __bf16 v16bf;
typedef __attribute__((ext_vector_type(8)))  __bf16 v8bf;
typedef __attribute__((ext_vector_type(8)))  float  v8f;

static __device__ __forceinline__ v8f wmma_bf16(v16bf a, v16bf b, v8f c) {
  // D = A(16x32 bf16) * B(32x16 bf16) + C(16x16 f32)
  return __builtin_amdgcn_wmma_f32_16x16x32_bf16(
      /*neg_a=*/false, a, /*neg_b=*/false, b,
      /*c_mod=*/(short)0, c, /*reuse_a=*/false, /*reuse_b=*/false);
}

static __device__ __forceinline__ v16bf cat8(v8bf lo, v8bf hi) {
  return __builtin_shufflevector(lo, hi, 0, 1, 2, 3, 4, 5, 6, 7,
                                 8, 9, 10, 11, 12, 13, 14, 15);
}

// ---- wide fragment gathers (layouts per CDNA5 ISA §7.12.2) ----
// A-matrix 16x32 bf16: lane l holds row M=l&15; element e -> K index:
//   K = (e>>3)*16 + (l>>4)*8 + (e&7)
static __device__ __forceinline__ v16bf load_A_frag_w(const __bf16* p, int ld,
                                                      int lane) {
  const int m  = lane & 15;
  const int kh = lane >> 4;
  const __bf16* q = p + m * ld + (kh << 3);
  const v8bf lo = *(const v8bf*)(q);
  const v8bf hi = *(const v8bf*)(q + 16);
  return cat8(lo, hi);
}

// A-fragment with only K=0..15 valid (upper half zero).
static __device__ __forceinline__ v16bf load_A_frag_k16(const __bf16* p, int ld,
                                                        int lane) {
  const int m  = lane & 15;
  const int kh = lane >> 4;
  const v8bf lo = *(const v8bf*)(p + m * ld + (kh << 3));
  const v8bf z  = {};
  return cat8(lo, z);
}

// B-fragment of M^T where memory holds M as (N x K) row-major:
//   element e -> K = (e>>3)*16 + (l>>4)*8 + (e&7).
static __device__ __forceinline__ v16bf load_BT_frag_w(const __bf16* p, int ld,
                                                       int lane) {
  const int nn = lane & 15;
  const int kh = lane >> 4;
  const __bf16* q = p + nn * ld + (kh << 3);
  const v8bf lo = *(const v8bf*)(q);
  const v8bf hi = *(const v8bf*)(q + 16);
  return cat8(lo, hi);
}

// Same but only K=0..15 valid: upper 8 elements are zero.
static __device__ __forceinline__ v16bf load_BT_frag_k16(const __bf16* p,
                                                         int ld, int lane) {
  const int nn = lane & 15;
  const int kh = lane >> 4;
  const __bf16* q = p + nn * ld + (kh << 3);
  const v8bf lo = *(const v8bf*)(q);
  const v8bf z  = {};
  return cat8(lo, z);
}

// B-fragment gathered from a row-major f32 matrix in global (pack only).
static __device__ __forceinline__ v16bf load_B_frag_g(const float* p, int ld,
                                                      int lane) {
  const int nn = lane & 15;
  const int kh = lane >> 4;
  v16bf b;
#pragma unroll
  for (int e = 0; e < 16; ++e) {
    const int kk = ((e >> 3) << 4) + (kh << 3) + (e & 7);
    b[e] = (__bf16)p[kk * ld + nn];
  }
  return b;
}

// A-fragment of M^T where global f32 memory holds M as (K x M) row-major:
//   A[m][kk] = p[kk*ld + m]   (used for Wq^T / Wk^T packing)
static __device__ __forceinline__ v16bf load_A_frag_gT(const float* p, int ld,
                                                       int lane) {
  const int m  = lane & 15;
  const int kh = lane >> 4;
  v16bf a;
#pragma unroll
  for (int e = 0; e < 16; ++e) {
    const int kk = ((e >> 3) << 4) + (kh << 3) + (e & 7);
    a[e] = (__bf16)p[kk * ld + m];
  }
  return a;
}

// Zero the pad rows (25..31) of a 32x64 LDS tile (contiguous 1600..2047).
static __device__ __forceinline__ void zero_x_pad(__bf16* Xb, int lane) {
#pragma unroll
  for (int it = 0; it < 14; ++it) Xb[1600 + it * 32 + lane] = (__bf16)0.0f;
}

// Stage X_t (25x64 f32 -> bf16) into a pre-pad-zeroed 32x64 LDS tile.
// 32-thread version (used by agg_kernel).
static __device__ __forceinline__ void stage_x(const float* xt, __bf16* Xb,
                                               int lane) {
#pragma unroll
  for (int it = 0; it < 12; ++it) {
    const int c4  = it * 32 + lane;      // 0..383 -> rows 0..23
    const int row = c4 >> 4;
    const int col = (c4 & 15) << 2;
    const float4 v = *(const float4*)(xt + row * C_ + col);
    __bf16* dst = &Xb[row * C_ + col];
    dst[0] = (__bf16)v.x; dst[1] = (__bf16)v.y;
    dst[2] = (__bf16)v.z; dst[3] = (__bf16)v.w;
  }
  if (lane < 16) {                       // chunks 384..399 -> row 24
    const int col = lane << 2;
    const float4 v = *(const float4*)(xt + 24 * C_ + col);
    __bf16* dst = &Xb[24 * C_ + col];
    dst[0] = (__bf16)v.x; dst[1] = (__bf16)v.y;
    dst[2] = (__bf16)v.z; dst[3] = (__bf16)v.w;
  }
}

// 96-thread version (attn_part): 400 float4 chunks = 4 full rounds + tail.
static __device__ __forceinline__ void stage_x96(const float* xt, __bf16* Xb,
                                                 int tid) {
#pragma unroll
  for (int it = 0; it < 4; ++it) {
    const int c4  = it * 96 + tid;       // 0..383
    const int row = c4 >> 4;
    const int col = (c4 & 15) << 2;
    const float4 v = *(const float4*)(xt + row * C_ + col);
    __bf16* dst = &Xb[row * C_ + col];
    dst[0] = (__bf16)v.x; dst[1] = (__bf16)v.y;
    dst[2] = (__bf16)v.z; dst[3] = (__bf16)v.w;
  }
  if (tid < 16) {                        // chunks 384..399 -> row 24
    const int col = tid << 2;
    const float4 v = *(const float4*)(xt + 24 * C_ + col);
    __bf16* dst = &Xb[24 * C_ + col];
    dst[0] = (__bf16)v.x; dst[1] = (__bf16)v.y;
    dst[2] = (__bf16)v.z; dst[3] = (__bf16)v.w;
  }
}

// =====================================================================
// Kernel 0a: one-time repack. W -> B-frags; Wq^T / Wk^T -> A-frags.
// =====================================================================
__global__ __launch_bounds__(32)
void pack_weights(const float* __restrict__ W, const float* __restrict__ Wq,
                  const float* __restrict__ Wk, __bf16* __restrict__ pk) {
  const int lane = threadIdx.x;
  const int bx = blockIdx.x;
  v16bf frag;
  int dstoff;
  if (bx < 24) {                       // W: fid = nt*2 + kc (B-fragment)
    const int nt = bx >> 1, kc = bx & 1;
    frag = load_B_frag_g(W + kc * 32 * (K_ * F_) + nt * 16, K_ * F_, lane);
    dstoff = WPACK_OFF + bx * 512;
  } else if (bx < 30) {                // Wq^T: fid = k*2 + kc (A-fragment)
    const int id = bx - 24, k = id >> 1, kc = id & 1;
    frag = load_A_frag_gT(Wq + (size_t)k * C_ * I_ + kc * 32 * I_, I_, lane);
    dstoff = WQPACK_OFF + id * 512;
  } else {                             // Wk^T
    const int id = bx - 30, k = id >> 1, kc = id & 1;
    frag = load_A_frag_gT(Wk + (size_t)k * C_ * I_ + kc * 32 * I_, I_, lane);
    dstoff = WKPACK_OFF + id * 512;
  }
  *(v16bf*)(pk + dstoff + lane * 16) = frag;
}

// =====================================================================
// Kernel 0b: zero Sws + Cws (202752 floats exactly). grid = 198 x 256.
// =====================================================================
__global__ __launch_bounds__(256)
void zero_s(float* __restrict__ Sws) {
  const size_t i = (size_t)blockIdx.x * 256 + threadIdx.x;
  *(float4*)(Sws + i * 4) = make_float4(0.f, 0.f, 0.f, 0.f);
}

// =====================================================================
// Kernel 1a: partial attention logits. grid = N_*CH_ blocks x 96 threads
// (3 waves, one head per wave; X staged cooperatively once per t).
// Bias-free Q/K; per-column bias term accumulated separately.
// Q^T = Wq^T * X^T so the D-tile is already row-major [v][i]: wide stores.
// =====================================================================
__global__ __launch_bounds__(96)
void attn_part_kernel(const float* __restrict__ x,
                      const float* __restrict__ bq,
                      const __bf16* __restrict__ pk,
                      float* __restrict__ Sws) {
  const int tid  = threadIdx.x;
  const int kg   = tid >> 5;          // wave id = head index
  const int lane = tid & 31;
  const int n  = blockIdx.x / CH_;
  const int t0 = (blockIdx.x % CH_) * TCH_;
  const int nn = lane & 15;
  const int kh = lane >> 4;

  __shared__ __attribute__((aligned(16))) __bf16 Xb[32 * 64];      // 4 KB
  __shared__ __attribute__((aligned(16))) __bf16 Qb[K_][32 * 16];  // 3 KB
  __shared__ __attribute__((aligned(16))) __bf16 Kb[K_][32 * 16];  // 3 KB
  __shared__ __attribute__((aligned(16))) __bf16 Wf[12 * 512];     // 12 KB

  // Stage all Wq^T/Wk^T A-fragments (contiguous 6144 bf16) to LDS.
#pragma unroll
  for (int it = 0; it < 8; ++it) {
    const int idx = it * 96 + tid;       // 768 8-element chunks
    *(v8bf*)(Wf + idx * 8) = *(const v8bf*)(pk + WQPACK_OFF + idx * 8);
  }
  // This wave's bq slice: rows i = kh*8 + r of its D-tiles.
  v8f bq8;
  {
    const float* bp = bq + kg * I_ + (kh << 3);
    const float4 b0 = *(const float4*)(bp);
    const float4 b1 = *(const float4*)(bp + 4);
    bq8[0] = b0.x; bq8[1] = b0.y; bq8[2] = b0.z; bq8[3] = b0.w;
    bq8[4] = b1.x; bq8[5] = b1.y; bq8[6] = b1.z; bq8[7] = b1.w;
  }
  // Zero X pad rows (448 contiguous bf16 at 1600): 4 full 96-wide rounds + 64.
#pragma unroll
  for (int it = 0; it < 4; ++it) Xb[1600 + it * 96 + tid] = (__bf16)0.0f;
  if (tid < 64) Xb[1600 + 384 + tid] = (__bf16)0.0f;

  const v8f zero = {};
  v8f S[2][2];
  S[0][0] = zero; S[0][1] = zero; S[1][0] = zero; S[1][1] = zero;
  float cpart[2] = {0.f, 0.f};

  __bf16* Qw = &Qb[kg][0];
  __bf16* Kw = &Kb[kg][0];

  const float* xbase = x + ((size_t)n * T_) * V_ * C_;  // t stride = V_*C_
  for (int t = t0; t < t0 + TCH_; ++t) {
    const float* xt = xbase + (size_t)t * (V_ * C_);
    if (t + 1 < T_ && tid < 32)
      __builtin_prefetch(xt + V_ * C_ + tid * 50, 0, 3);
    stage_x96(xt, Xb, tid);
    __syncthreads();

    // B-fragments of X^T: contiguous wide gathers from row-major Xb.
    v16bf xbT[2][2];
#pragma unroll
    for (int nt = 0; nt < 2; ++nt)
#pragma unroll
      for (int kc = 0; kc < 2; ++kc)
        xbT[nt][kc] = load_BT_frag_w(&Xb[(nt * 16) * C_ + kc * 32], C_, lane);

    // ---- Q pass ----
    {
      const v16bf wA0 = *(const v16bf*)(Wf + (kg * 2 + 0) * 512 + lane * 16);
      const v16bf wA1 = *(const v16bf*)(Wf + (kg * 2 + 1) * 512 + lane * 16);
      v8f acc[2];
      acc[0] = wmma_bf16(wA0, xbT[0][0], zero);
      acc[1] = wmma_bf16(wA0, xbT[1][0], zero);
      acc[0] = wmma_bf16(wA1, xbT[0][1], acc[0]);
      acc[1] = wmma_bf16(wA1, xbT[1][1], acc[1]);
#pragma unroll
      for (int nt = 0; nt < 2; ++nt) {
        v8bf qv;
#pragma unroll
        for (int r = 0; r < 8; ++r) qv[r] = (__bf16)acc[nt][r];
        // D-tile: lane col v = nt*16+nn fixed, rows i = kh*8+r contiguous.
        *(v8bf*)(&Qw[(nt * 16 + nn) * 16 + (kh << 3)]) = qv;   // 16B store
      }
    }
    // ---- K pass ----
    {
      const v16bf wA0 =
          *(const v16bf*)(Wf + 3072 + (kg * 2 + 0) * 512 + lane * 16);
      const v16bf wA1 =
          *(const v16bf*)(Wf + 3072 + (kg * 2 + 1) * 512 + lane * 16);
      v8f acc[2];
      acc[0] = wmma_bf16(wA0, xbT[0][0], zero);
      acc[1] = wmma_bf16(wA0, xbT[1][0], zero);
      acc[0] = wmma_bf16(wA1, xbT[0][1], acc[0]);
      acc[1] = wmma_bf16(wA1, xbT[1][1], acc[1]);
#pragma unroll
      for (int nt = 0; nt < 2; ++nt) {
        v8bf kv;
#pragma unroll
        for (int r = 0; r < 8; ++r) {
          // Column bias term: c[w] += sum_i K^T[i][w] * bq[i]
          cpart[nt] += acc[nt][r] * bq8[r];
          kv[r] = (__bf16)acc[nt][r];
        }
        *(v8bf*)(&Kw[(nt * 16 + nn) * 16 + (kh << 3)]) = kv;   // 16B store
      }
    }

    // S[v][w] += sum_i Q[v][i] * K[w][i]  (K-dim padded 16 -> 32).
    // Per-wave private Qb/Kb: LDS ops are in-order within a wave, so no
    // barrier is needed between the stores above and these gathers.
    v16bf qa[2], kbT[2];
#pragma unroll
    for (int mt = 0; mt < 2; ++mt)
      qa[mt] = load_A_frag_k16(&Qw[mt * 16 * 16], 16, lane);
#pragma unroll
    for (int nt = 0; nt < 2; ++nt)
      kbT[nt] = load_BT_frag_k16(&Kw[nt * 16 * 16], 16, lane);
#pragma unroll
    for (int mt = 0; mt < 2; ++mt)
#pragma unroll
      for (int nt = 0; nt < 2; ++nt)
        S[mt][nt] = wmma_bf16(qa[mt], kbT[nt], S[mt][nt]);
    __syncthreads();   // all waves done reading Xb before next stage_x
  }

  // Reduce partial logit tiles + column bias terms with f32 atomics.
  float* dst = Sws + (((size_t)kg * N_ + n) << 10);
#pragma unroll
  for (int mt = 0; mt < 2; ++mt)
#pragma unroll
    for (int nt = 0; nt < 2; ++nt)
#pragma unroll
      for (int r = 0; r < 8; ++r) {
        const int row = mt * 16 + r + (kh << 3);
        atomicAdd(dst + row * 32 + nt * 16 + nn, S[mt][nt][r]);
      }
  float* Cws = Sws + SWS_FLOATS;
#pragma unroll
  for (int nt = 0; nt < 2; ++nt)
    atomicAdd(Cws + ((size_t)kg * N_ + n) * 32 + nt * 16 + nn, cpart[nt]);
}

// =====================================================================
// Kernel 1b: softmax (+ column bias) + A-add + fragment pack. grid = K_*N_.
// =====================================================================
__global__ __launch_bounds__(32)
void attn_final_kernel(const float* __restrict__ Sws,
                       const float* __restrict__ A,
                       __bf16* __restrict__ pk) {
  const int lane = threadIdx.x;
  const int kg = blockIdx.x / N_;
  const int n  = blockIdx.x % N_;

  __shared__ __attribute__((aligned(16))) float Sb[32 * 32];
  __shared__ __attribute__((aligned(16))) float Cb[32];

  const float* src = Sws + (((size_t)kg * N_ + n) << 10);
#pragma unroll
  for (int it = 0; it < 8; ++it) {
    const int idx = it * 32 + lane;      // 256 float4 chunks
    *(float4*)(&Sb[idx * 4]) = *(const float4*)(src + idx * 4);
  }
  Cb[lane] = Sws[SWS_FLOATS + ((size_t)kg * N_ + n) * 32 + lane];
  __syncthreads();

  // Softmax over w per row v (logit = S + c[w]), add static A.
  float mxs = 0.f, inv = 0.f;
  if (lane < V_) {
    float mx = -3.0e38f;
    for (int w = 0; w < V_; ++w)
      mx = fmaxf(mx, Sb[lane * 32 + w] + Cb[w]);
    mxs = mx * SCALE_;
    float sum = 0.f;
    for (int w = 0; w < V_; ++w)
      sum += __expf((Sb[lane * 32 + w] + Cb[w]) * SCALE_ - mxs);
    inv = 1.0f / sum;
  }
  for (int w = 0; w < 32; ++w) {
    float val = 0.f;
    if (lane < V_ && w < V_)
      val = A[((size_t)kg * V_ + lane) * V_ + w] +
            __expf((Sb[lane * 32 + w] + Cb[w]) * SCALE_ - mxs) * inv;
    Sb[lane * 32 + w] = val;   // lane reads its own row before overwriting
  }
  __syncthreads();

  // Emit bf16 A-fragments of A_adapt^T ([w][v]) for the aggregation GEMM.
#pragma unroll
  for (int mt = 0; mt < 2; ++mt) {
    const int w_ = mt * 16 + (lane & 15);
    const int kh = lane >> 4;
    v16bf af;
#pragma unroll
    for (int e = 0; e < 16; ++e) {
      const int kk = ((e >> 3) << 4) + (kh << 3) + (e & 7);   // v index
      af[e] = (__bf16)Sb[kk * 32 + w_];
    }
    *(v16bf*)(pk + APACK_OFF + (((size_t)kg * N_ + n) * 2 + mt) * 512 +
              lane * 16) = af;
  }
}

// =====================================================================
// Kernel 2: per (n,t) fused conv + aggregation
//   H = X_t*W + b (25x192) ; y[w,c] = sum_k sum_v Aad[k,n,v,w] * H[v,k*64+c]
// grid = N_*T_ blocks, 32 threads (1 wave)
// =====================================================================
__global__ __launch_bounds__(32)
void agg_kernel(const float* __restrict__ x, const float* __restrict__ b,
                const __bf16* __restrict__ pk, float* __restrict__ y) {
  const int lane = threadIdx.x;
  const int n = blockIdx.x / T_;
  const int t = blockIdx.x % T_;

  __shared__ __attribute__((aligned(16))) __bf16 Xb[32 * 64];    // 4 KB
  __shared__ __attribute__((aligned(16))) __bf16 HbT[192 * 32];  // 12 KB

  const float* xt = x + (((size_t)n * T_ + t) * V_) * C_;
  zero_x_pad(Xb, lane);
  stage_x(xt, Xb, lane);
  __syncthreads();

  v16bf xa[2][2];
#pragma unroll
  for (int mt = 0; mt < 2; ++mt)
#pragma unroll
    for (int kc = 0; kc < 2; ++kc)
      xa[mt][kc] = load_A_frag_w(&Xb[mt * 16 * C_ + kc * 32], C_, lane);

  const v8f zero = {};
  const int kh = lane >> 4;
  const int nn = lane & 15;

  // H = X*W + b -> HbT[c][v], one 16-wide column block at a time.
  for (int nt = 0; nt < 12; ++nt) {
    const v16bf wb0 =
        *(const v16bf*)(pk + WPACK_OFF + (nt * 2 + 0) * 512 + lane * 16);
    const v16bf wb1 =
        *(const v16bf*)(pk + WPACK_OFF + (nt * 2 + 1) * 512 + lane * 16);
    const int cc = nt * 16 + nn;           // this lane's output column
    const float bv = b[cc];
#pragma unroll
    for (int mt = 0; mt < 2; ++mt) {
      v8f acc = zero;
      acc = wmma_bf16(xa[mt][0], wb0, acc);
      acc = wmma_bf16(xa[mt][1], wb1, acc);
      v8bf hv;
#pragma unroll
      for (int r = 0; r < 8; ++r) {
        const int row = mt * 16 + (kh << 3) + r;
        const float val = (row < V_) ? (acc[r] + bv) : 0.0f;  // keep pads 0
        hv[r] = (__bf16)val;
      }
      *(v8bf*)(&HbT[cc * 32 + mt * 16 + (kh << 3)]) = hv;     // 16B store
    }
  }
  __syncthreads();

  // y(w,c) = sum_k Aad_k^T (25x25) * H_k (25x64), contraction padded to 32.
  v8f ya[2][4];
#pragma unroll
  for (int mt = 0; mt < 2; ++mt)
#pragma unroll
    for (int nt = 0; nt < 4; ++nt) ya[mt][nt] = zero;

#pragma unroll
  for (int kg = 0; kg < K_; ++kg) {
    v16bf aa[2];
#pragma unroll
    for (int mt = 0; mt < 2; ++mt)
      aa[mt] = *(const v16bf*)(pk + APACK_OFF +
                               (((size_t)kg * N_ + n) * 2 + mt) * 512 +
                               lane * 16);
#pragma unroll
    for (int nt = 0; nt < 4; ++nt) {
      const v16bf hb = load_BT_frag_w(&HbT[(kg * 64 + nt * 16) * 32], 32, lane);
      ya[0][nt] = wmma_bf16(aa[0], hb, ya[0][nt]);
      ya[1][nt] = wmma_bf16(aa[1], hb, ya[1][nt]);
    }
  }

  // Store y[n,t,w,c]
  float* yt = y + (((size_t)n * T_ + t) * V_) * C_;
#pragma unroll
  for (int mt = 0; mt < 2; ++mt)
#pragma unroll
    for (int r = 0; r < 8; ++r) {
      const int w_ = mt * 16 + r + (kh << 3);
      if (w_ < V_) {
#pragma unroll
        for (int nt = 0; nt < 4; ++nt)
          yt[w_ * C_ + nt * 16 + nn] = ya[mt][nt][r];
      }
    }
}

extern "C" void kernel_launch(void* const* d_in, const int* in_sizes, int n_in,
                              void* d_out, int out_size, void* d_ws, size_t ws_size,
                              hipStream_t stream) {
  const float* x  = (const float*)d_in[0];
  const float* A  = (const float*)d_in[1];
  const float* W  = (const float*)d_in[2];
  const float* b  = (const float*)d_in[3];
  const float* Wq = (const float*)d_in[4];
  const float* bq = (const float*)d_in[5];
  const float* Wk = (const float*)d_in[6];
  float*  y   = (float*)d_out;
  float*  Sws = (float*)d_ws;                                   // 811 KB
  __bf16* pk  = (__bf16*)((char*)d_ws + (size_t)ZERO_FLOATS * 4);  // 430 KB

  pack_weights<<<36, 32, 0, stream>>>(W, Wq, Wk, pk);
  zero_s<<<198, 256, 0, stream>>>(Sws);
  attn_part_kernel<<<N_ * CH_, 96, 0, stream>>>(x, bq, pk, Sws);
  attn_final_kernel<<<K_ * N_, 32, 0, stream>>>(Sws, A, pk);
  agg_kernel<<<N_ * T_, 32, 0, stream>>>(x, b, pk, y);
}
